// FPInitializerNew_53970559042024
// MI455X (gfx1250) — compile-verified, matrix operand order checked by reading, not run on hardware
//
#include <hip/hip_runtime.h>
#include <hip/hip_bf16.h>

typedef __attribute__((ext_vector_type(16))) _Float16 v16h;
typedef __attribute__((ext_vector_type(8)))  _Float16 v8h;
typedef __attribute__((ext_vector_type(8)))  float    v8f;

// ---------------------------------------------------------------------------
// Shapes
//   B=512, A=64, Bo=96, D=6, AF=39, BF=10, FP=150
//   atom rows = 512*64 = 32768        (39 -> 150 -> 150)
//   bond rows = 512*96 = 49152        (10 -> 150 -> 150)
//   nei  rows = 512*64*6 = 196608     (450 -> 150 -> 150)
// All K dims padded to multiples of 32 (f16 WMMA K), N padded 150 -> 160.
// ---------------------------------------------------------------------------

#define EPS 1e-6f
#define NT 10   // 160 / 16 output tiles

__device__ __forceinline__ v16h cat16(v8h lo, v8h hi) {
    v16h r;
#pragma unroll
    for (int i = 0; i < 8; ++i) { r[i] = lo[i]; r[8 + i] = hi[i]; }
    return r;
}

// One wave computes a 16-row x 160-col tile: acc += A(16 x KT*32) * W(KT*32 x 160)
// sA: f16 row-major [16][KT*32] in LDS (wave-local base)
// wf: weights pre-swizzled into B-fragment layout:
//     frag(kt,nt) = 512 halves at ((kt*NT+nt)*32 + lane)*16; element e -> K offset
//     (e&7) + (e>=8 ? 16:0) + (lane>=16 ? 8:0), col = nt*16 + (lane&15).
template <int KT>
__device__ __forceinline__ void gemm16(const _Float16* sA, const _Float16* __restrict__ wf,
                                       int lane, v8f (&acc)[NT]) {
    const int stride = KT * 32;
    const int row    = lane & 15;
    const int hsel   = (lane & 16) ? 8 : 0;
    for (int kt = 0; kt < KT; ++kt) {
        const _Float16* ap = sA + row * stride + kt * 32 + hsel;
        v16h a = cat16(*(const v8h*)ap, *(const v8h*)(ap + 16));
#pragma unroll
        for (int nt = 0; nt < NT; ++nt) {
            const _Float16* bp = wf + (((size_t)(kt * NT + nt)) * 32 + lane) * 16;
            v16h b = cat16(*(const v8h*)bp, *(const v8h*)(bp + 8));
            acc[nt] = __builtin_amdgcn_wmma_f32_16x16x32_f16(
                false, a, false, b, (short)0, acc[nt], false, false);
        }
    }
}

// bias + ReLU, write f16 hidden tile [16][160] to LDS (wave-local base).
// C/D layout: lane<16 -> N=lane, M=vgpr j; lane>=16 -> same N, M=j+8.
__device__ __forceinline__ void epi_hidden(const v8f (&acc)[NT], const float* __restrict__ bias,
                                           _Float16* hid, int lane) {
    const int c0 = lane & 15;
    const int mb = (lane & 16) ? 8 : 0;
#pragma unroll
    for (int nt = 0; nt < NT; ++nt) {
        const int col = nt * 16 + c0;
        const float bb = bias[col];
#pragma unroll
        for (int j = 0; j < 8; ++j) {
            float v = fmaxf(acc[nt][j] + bb, 0.f);
            hid[(mb + j) * 160 + col] = (_Float16)v;
        }
    }
}

// bias + ReLU, store f32 rows [rowbase..rowbase+15] x [0..149] to global.
__device__ __forceinline__ void epi_out(const v8f (&acc)[NT], const float* __restrict__ bias,
                                        float* __restrict__ out, long rowbase, int lane) {
    const int c0 = lane & 15;
    const int mb = (lane & 16) ? 8 : 0;
#pragma unroll
    for (int nt = 0; nt < NT; ++nt) {
        const int col = nt * 16 + c0;
        if (col < 150) {
            const float bb = bias[col];
#pragma unroll
            for (int j = 0; j < 8; ++j) {
                float v = fmaxf(acc[nt][j] + bb, 0.f);
                out[(rowbase + mb + j) * 150 + col] = v;
            }
        }
    }
}

// ---------------------------------------------------------------------------
// BN fold + weight swizzle:  W' = W * gamma*rsqrt(var+eps), b' = (b-mean)*s+beta
// Writes weights in WMMA B-fragment layout (f16), bias padded to 160 (f32).
// ---------------------------------------------------------------------------
__global__ void fold_kernel(const float* __restrict__ W, const float* __restrict__ b,
                            const float* __restrict__ beta, const float* __restrict__ gamma,
                            const float* __restrict__ mean, const float* __restrict__ var,
                            _Float16* __restrict__ wfrag, float* __restrict__ bias,
                            int CIN, int nhalves) {
    const int COUT = 150;
    int gid = blockIdx.x * blockDim.x + threadIdx.x;
    if (gid < 160) {
        float v = 0.f;
        if (gid < COUT) {
            float s = gamma[gid] * rsqrtf(var[gid] + EPS);
            v = (b[gid] - mean[gid]) * s + beta[gid];
        }
        bias[gid] = v;
    }
    if (gid >= nhalves) return;
    int frag = gid >> 9;          // 512 halves per fragment
    int lane = (gid >> 4) & 31;
    int e    = gid & 15;
    int kt   = frag / NT;
    int nt   = frag - kt * NT;
    int col  = nt * 16 + (lane & 15);
    int k    = kt * 32 + (e & 7) + ((e & 8) ? 16 : 0) + ((lane & 16) ? 8 : 0);
    float v = 0.f;
    if (col < COUT && k < CIN) {
        float s = gamma[col] * rsqrtf(var[col] + EPS);
        v = W[col * CIN + k] * s;
    }
    wfrag[gid] = (_Float16)v;
}

// ---------------------------------------------------------------------------
// Dense two-layer block: x[rows][CIN] -> relu(bn(lin)) -> relu(bn(lin)) -> out
// 4 waves per WG, 64 rows per WG, one 16-row tile per wave.
// ---------------------------------------------------------------------------
template <int CIN, int K1T>
__global__ __launch_bounds__(128) void linbn_block_kernel(
    const float* __restrict__ x,
    const _Float16* __restrict__ wf1, const float* __restrict__ bias1,
    const _Float16* __restrict__ wf2, const float* __restrict__ bias2,
    float* __restrict__ out) {
    __shared__ _Float16 sIn[64 * K1T * 32];
    __shared__ _Float16 sHid[64 * 160];
    const int tid  = threadIdx.x;
    const int lane = tid & 31;
    const int wave = tid >> 5;
    const long rowbase = (long)blockIdx.x * 64;

    // Stage input rows as f16, zero-padded to K1T*32
    for (int i = tid; i < 64 * K1T * 32; i += 128) {
        int r = i / (K1T * 32), k = i - r * (K1T * 32);
        float v = (k < CIN) ? x[(rowbase + r) * CIN + k] : 0.f;
        sIn[i] = (_Float16)v;
    }
    __syncthreads();

    v8f acc[NT];
#pragma unroll
    for (int nt = 0; nt < NT; ++nt) { v8f z = {}; acc[nt] = z; }
    gemm16<K1T>(sIn + wave * 16 * K1T * 32, wf1, lane, acc);
    epi_hidden(acc, bias1, sHid + wave * 16 * 160, lane);
    __syncthreads();

    v8f acc2[NT];
#pragma unroll
    for (int nt = 0; nt < NT; ++nt) { v8f z = {}; acc2[nt] = z; }
    gemm16<5>(sHid + wave * 16 * 160, wf2, lane, acc2);
    epi_out(acc2, bias2, out, rowbase + wave * 16, lane);
}

// ---------------------------------------------------------------------------
// Neighbor block: gather atom/bond FP rows, mix, concat (450 -> pad 480),
// then 450->150->150. 2 waves per WG, 32 rows per WG.
// ---------------------------------------------------------------------------
__global__ __launch_bounds__(64) void neighbor_kernel(
    const float* __restrict__ atomFP, const float* __restrict__ bondFP,
    const int* __restrict__ anl, const int* __restrict__ bnl,
    const _Float16* __restrict__ wf1, const float* __restrict__ bias1,
    const _Float16* __restrict__ wf2, const float* __restrict__ bias2,
    float* __restrict__ out) {
    __shared__ _Float16 sIn[32 * 480];
    __shared__ _Float16 sHid[32 * 160];
    __shared__ int sIa[32], sIb[32];
    const int tid  = threadIdx.x;
    const int lane = tid & 31;
    const int wave = tid >> 5;
    const long rowbase = (long)blockIdx.x * 32;

    if (tid < 32) {
        long row = rowbase + tid;
        int batch = (int)(row / 384);          // 64*6
        int rem   = (int)(row - (long)batch * 384);
        int a = rem / 6, d = rem - (rem / 6) * 6;
        sIa[tid] = anl[((size_t)batch * 64 + a) * 6 + d];
        sIb[tid] = bnl[((size_t)batch * 64 + a) * 6 + d];
    }
    __syncthreads();

    // Gather + mixture -> f16 LDS tile [32][480]: [a | b | a+b-a*b | pad]
    for (int i = tid; i < 32 * 150; i += 64) {
        int r = i / 150, k = i - r * 150;
        int batch = (int)((rowbase + r) / 384);
        float av = atomFP[((size_t)batch * 64 + sIa[r]) * 150 + k];
        float bv = bondFP[((size_t)batch * 96 + sIb[r]) * 150 + k];
        float mx = av + bv - av * bv;
        _Float16* p = sIn + r * 480;
        p[k]       = (_Float16)av;
        p[150 + k] = (_Float16)bv;
        p[300 + k] = (_Float16)mx;
    }
    for (int i = tid; i < 32 * 30; i += 64) {   // zero K-pad 450..479
        int r = i / 30, c = i - r * 30;
        sIn[r * 480 + 450 + c] = (_Float16)0.f;
    }
    __syncthreads();

    v8f acc[NT];
#pragma unroll
    for (int nt = 0; nt < NT; ++nt) { v8f z = {}; acc[nt] = z; }
    gemm16<15>(sIn + wave * 16 * 480, wf1, lane, acc);
    epi_hidden(acc, bias1, sHid + wave * 16 * 160, lane);
    __syncthreads();

    v8f acc2[NT];
#pragma unroll
    for (int nt = 0; nt < NT; ++nt) { v8f z = {}; acc2[nt] = z; }
    gemm16<5>(sHid + wave * 16 * 160, wf2, lane, acc2);
    epi_out(acc2, bias2, out, rowbase + wave * 16, lane);
}

// ---------------------------------------------------------------------------
// Host launch
// Inputs (JAX sorted-key flatten):
//   0 atom_features [512,64,39] f32
//   1 bond_features [512,96,10] f32
//   2 atom_neighbor_list [512,64,6] i32
//   3 bond_neighbor_list [512,64,6] i32
//   4.. params: atom.l1, atom.l2, bond.l1, bond.l2, nei.l1, nei.l2
//       each: W, b, beta, gamma, mean, var
// Output: atom_FP [512*64*150] ++ neighbor_FP [512*64*6*150]  (f32)
// ---------------------------------------------------------------------------
extern "C" void kernel_launch(void* const* d_in, const int* in_sizes, int n_in,
                              void* d_out, int out_size, void* d_ws, size_t ws_size,
                              hipStream_t stream) {
    const float* af  = (const float*)d_in[0];
    const float* bfi = (const float*)d_in[1];
    const int*   anl = (const int*)d_in[2];
    const int*   bnl = (const int*)d_in[3];

    auto P = [&](int layer, int t) { return (const float*)d_in[4 + layer * 6 + t]; };
    // layer order: 0=atom.l1 1=atom.l2 2=bond.l1 3=bond.l2 4=nei.l1 5=nei.l2

    // Workspace layout: [bondFP f32][biases 6*160 f32][wfrags f16]
    float* bondFP = (float*)d_ws;
    const size_t bondElems = (size_t)512 * 96 * 150;
    float* biases = bondFP + bondElems;
    _Float16* wfrags = (_Float16*)(biases + 6 * 160);

    static const size_t woff[6] = {0, 10240, 35840, 40960, 66560, 143360};
    static const int    cins[6] = {39, 150, 10, 150, 450, 150};
    static const int    kts[6]  = {2, 5, 1, 5, 15, 5};

    for (int l = 0; l < 6; ++l) {
        int nh = kts[l] * NT * 512;
        int blocks = (nh + 255) / 256;
        fold_kernel<<<blocks, 256, 0, stream>>>(
            P(l, 0), P(l, 1), P(l, 2), P(l, 3), P(l, 4), P(l, 5),
            wfrags + woff[l], biases + l * 160, cins[l], nh);
    }

    float* atomFP = (float*)d_out;                       // 512*64 rows
    float* neiFP  = atomFP + (size_t)512 * 64 * 150;     // 512*64*6 rows

    linbn_block_kernel<39, 2><<<512, 128, 0, stream>>>(
        af, wfrags + woff[0], biases + 0, wfrags + woff[1], biases + 160, atomFP);
    linbn_block_kernel<10, 1><<<768, 128, 0, stream>>>(
        bfi, wfrags + woff[2], biases + 320, wfrags + woff[3], biases + 480, bondFP);
    neighbor_kernel<<<6144, 64, 0, stream>>>(
        atomFP, bondFP, anl, bnl,
        wfrags + woff[4], biases + 640, wfrags + woff[5], biases + 800, neiFP);
}
